// Denoise_module_61220463837888
// MI455X (gfx1250) — matrix operand order repaired and checked by reading, not compiled, not run on hardware
//
#include <hip/hip_runtime.h>

// ---------------------------------------------------------------------------
// Problem constants (from the reference)
// ---------------------------------------------------------------------------
namespace cfg {
constexpr int B  = 32;
constexpr int L  = 512;
constexpr int N  = 16;
constexpr int H  = 768;
constexpr int NH = 4;
constexpr int DH = 192;   // H / NH
constexpr int LP = 528;   // L + N
constexpr int H2 = 1536;  // 2H
constexpr int H3 = 2304;  // 3H
constexpr int NTOP = 32;  // K+1 = L/N
}

// ---------------------------------------------------------------------------
// bf16 WMMA types / helpers (CDNA5 / gfx1250, wave32)
// ---------------------------------------------------------------------------
typedef __attribute__((ext_vector_type(16))) __bf16 v16bf;
typedef __attribute__((ext_vector_type(8)))  float  v8f;

__device__ __forceinline__ unsigned short f2bf(float f) {
  unsigned int u = __float_as_uint(f);
  u += 0x7FFFu + ((u >> 16) & 1u);          // round-to-nearest-even
  return (unsigned short)(u >> 16);
}

__device__ __forceinline__ uint2 pack4bf(float4 f) {
  uint2 r;
  r.x = (unsigned)f2bf(f.x) | ((unsigned)f2bf(f.y) << 16);
  r.y = (unsigned)f2bf(f.z) | ((unsigned)f2bf(f.w) << 16);
  return r;
}

union FragB16 { v16bf v; unsigned int u[8]; };

// ---------------------------------------------------------------------------
// Generic batched tiled WMMA GEMM:  C = A(MxK) * op(B) (+ bias)
//   TRANSB=false : B is KxN row-major -> C = A*B
//   TRANSB=true  : B is NxK row-major -> C = A*B^T  (gram / q.k^T)
// fp32 in HBM, converted to bf16 while staging tiles into LDS.
// Block: 256 threads = 8 waves; block tile 128x64; wave tile 32x32 (2x2 WMMA).
// Double-buffered LDS, float4 global loads on interior tiles.
// Per-batch offsets:  off = (bb/nh)*stride + (bb%nh)*hstride.
// ---------------------------------------------------------------------------
template <bool TRANSB, bool HASBIAS>
__global__ __launch_bounds__(256) void wmma_gemm(
    const float* __restrict__ A, const float* __restrict__ Bm,
    const float* __restrict__ bias, float* __restrict__ C,
    int M, int N, int K, int lda, int ldb, int ldc,
    long sA, long hA, long sB, long hB, long sC, long hC, int nh)
{
  constexpr int TM = 128, TN = 64, TK = 32;
  __shared__ unsigned short As[2][TM][TK + 8];   // [buf][m][k]
  __shared__ unsigned short Bs[2][TN][TK + 8];   // [buf][n][k]  (B^T layout)

  const int bb = blockIdx.z;
  const float* Ab = A  + (long)(bb / nh) * sA + (long)(bb % nh) * hA;
  const float* Bb = Bm + (long)(bb / nh) * sB + (long)(bb % nh) * hB;
  float*       Cb = C  + (long)(bb / nh) * sC + (long)(bb % nh) * hC;

  const int tid  = threadIdx.x;
  const int mb   = blockIdx.y * TM;
  const int nb   = blockIdx.x * TN;
  const int lane = tid & 31;
  const int w    = tid >> 5;
  const int half = (lane >> 4) & 1;
  const int lq   = lane & 15;
  const int wm   = (w & 3) * 32;
  const int wn   = (w >> 2) * 32;

  const v8f vzero = {0.f, 0.f, 0.f, 0.f, 0.f, 0.f, 0.f, 0.f};
  v8f acc[2][2];
#pragma unroll
  for (int i = 0; i < 2; ++i)
#pragma unroll
    for (int j = 0; j < 2; ++j) acc[i][j] = vzero;

  const int ksteps = (K + TK - 1) / TK;
  float4 aR[4];   // staged A tile fragment (4 x float4 per thread)
  float4 bR[2];   // staged B tile fragment (2 x float4 per thread)

  // ---- stage tile kt from global into registers ----
  auto loadTiles = [&](int kt) {
    const int k0 = kt * TK;
    const bool fullK = (k0 + TK <= K);
    // A: 128x32 floats = 1024 float4; thread does 4
    const bool fastA = (mb + TM <= M) && fullK;
#pragma unroll
    for (int it = 0; it < 4; ++it) {
      const int i4 = it * 256 + tid;
      const int m = i4 >> 3, k4 = (i4 & 7) * 4;
      const int gm = mb + m, gk = k0 + k4;
      if (fastA) {
        aR[it] = *(const float4*)&Ab[(long)gm * lda + gk];
      } else {
        float4 f;
        f.x = (gm < M && gk + 0 < K) ? Ab[(long)gm * lda + gk + 0] : 0.f;
        f.y = (gm < M && gk + 1 < K) ? Ab[(long)gm * lda + gk + 1] : 0.f;
        f.z = (gm < M && gk + 2 < K) ? Ab[(long)gm * lda + gk + 2] : 0.f;
        f.w = (gm < M && gk + 3 < K) ? Ab[(long)gm * lda + gk + 3] : 0.f;
        aR[it] = f;
      }
    }
    // B: 64x32 = 512 float4; thread does 2
    const bool fastB = (nb + TN <= N) && fullK;
#pragma unroll
    for (int it = 0; it < 2; ++it) {
      const int i4 = it * 256 + tid;
      if (TRANSB) {
        const int n = i4 >> 3, k4 = (i4 & 7) * 4;
        const int gn = nb + n, gk = k0 + k4;
        if (fastB) {
          bR[it] = *(const float4*)&Bb[(long)gn * ldb + gk];
        } else {
          float4 f;
          f.x = (gn < N && gk + 0 < K) ? Bb[(long)gn * ldb + gk + 0] : 0.f;
          f.y = (gn < N && gk + 1 < K) ? Bb[(long)gn * ldb + gk + 1] : 0.f;
          f.z = (gn < N && gk + 2 < K) ? Bb[(long)gn * ldb + gk + 2] : 0.f;
          f.w = (gn < N && gk + 3 < K) ? Bb[(long)gn * ldb + gk + 3] : 0.f;
          bR[it] = f;
        }
      } else {
        const int k = i4 >> 4, n4 = (i4 & 15) * 4;
        const int gk = k0 + k, gn = nb + n4;
        if (fastB) {
          bR[it] = *(const float4*)&Bb[(long)gk * ldb + gn];
        } else {
          float4 f;
          f.x = (gk < K && gn + 0 < N) ? Bb[(long)gk * ldb + gn + 0] : 0.f;
          f.y = (gk < K && gn + 1 < N) ? Bb[(long)gk * ldb + gn + 1] : 0.f;
          f.z = (gk < K && gn + 2 < N) ? Bb[(long)gk * ldb + gn + 2] : 0.f;
          f.w = (gk < K && gn + 3 < N) ? Bb[(long)gk * ldb + gn + 3] : 0.f;
          bR[it] = f;
        }
      }
    }
  };

  // ---- write staged registers (bf16-converted) into LDS buffer ----
  auto storeTiles = [&](int buf) {
#pragma unroll
    for (int it = 0; it < 4; ++it) {
      const int i4 = it * 256 + tid;
      const int m = i4 >> 3, k4 = (i4 & 7) * 4;
      *(uint2*)&As[buf][m][k4] = pack4bf(aR[it]);
    }
#pragma unroll
    for (int it = 0; it < 2; ++it) {
      const int i4 = it * 256 + tid;
      if (TRANSB) {
        const int n = i4 >> 3, k4 = (i4 & 7) * 4;
        *(uint2*)&Bs[buf][n][k4] = pack4bf(bR[it]);
      } else {
        const int k = i4 >> 4, n4 = (i4 & 15) * 4;
        Bs[buf][n4 + 0][k] = f2bf(bR[it].x);
        Bs[buf][n4 + 1][k] = f2bf(bR[it].y);
        Bs[buf][n4 + 2][k] = f2bf(bR[it].z);
        Bs[buf][n4 + 3][k] = f2bf(bR[it].w);
      }
    }
  };

  // ---- WMMA on one LDS buffer (fragments per ISA 7.12.2 layouts) ----
  auto compute = [&](int buf) {
    FragB16 af[2], bfr[2];
#pragma unroll
    for (int ti = 0; ti < 2; ++ti) {
      const int m = wm + ti * 16 + lq;
#pragma unroll
      for (int i = 0; i < 8; ++i) {
        const int k = (i < 4) ? (half * 8 + 2 * i) : (16 + half * 8 + 2 * (i - 4));
        af[ti].u[i] = *(const unsigned int*)&As[buf][m][k];
      }
    }
#pragma unroll
    for (int tj = 0; tj < 2; ++tj) {
      const int n = wn + tj * 16 + lq;
#pragma unroll
      for (int i = 0; i < 8; ++i) {
        const int k = (i < 4) ? (half * 8 + 2 * i) : (16 + half * 8 + 2 * (i - 4));
        bfr[tj].u[i] = *(const unsigned int*)&Bs[buf][n][k];
      }
    }
#pragma unroll
    for (int ti = 0; ti < 2; ++ti)
#pragma unroll
      for (int tj = 0; tj < 2; ++tj)
        acc[ti][tj] = __builtin_amdgcn_wmma_f32_16x16x32_bf16(
            false, af[ti].v, false, bfr[tj].v, (short)0, acc[ti][tj], false, false);
  };

  // ---- software pipeline: stage kt+1 while computing kt ----
  loadTiles(0);
  storeTiles(0);
  __syncthreads();
  for (int kt = 0; kt < ksteps; ++kt) {
    const int cur = kt & 1;
    const bool more = (kt + 1 < ksteps);
    if (kt + 2 < ksteps) {  // speculative prefetch two tiles ahead
      __builtin_prefetch(&Ab[(long)(mb + (tid >> 1)) * lda + (kt + 2) * TK], 0, 3);
    }
    if (more) loadTiles(kt + 1);
    compute(cur);
    if (more) storeTiles(cur ^ 1);
    __syncthreads();
  }

  // ---- epilogue: C layout (row = tile + half*8 + r, col = tile + lq) ----
  const bool fullC = (mb + TM <= M) && (nb + TN <= N);
#pragma unroll
  for (int ti = 0; ti < 2; ++ti) {
#pragma unroll
    for (int tj = 0; tj < 2; ++tj) {
      const int cn = nb + wn + tj * 16 + lq;
#pragma unroll
      for (int r = 0; r < 8; ++r) {
        const int cm = mb + wm + ti * 16 + half * 8 + r;
        if (fullC || (cm < M && cn < N)) {
          float v = acc[ti][tj][r];
          if (HASBIAS) v += bias[cn];
          Cb[(long)cm * ldc + cn] = v;
        }
      }
    }
  }
}

// ---------------------------------------------------------------------------
// Wave-per-row softmax (wave32). Optional pre-scale and diagonal additive term
// (element m == row % diagMod gets += diagAdd). Rows contiguous, length Lrow.
// ---------------------------------------------------------------------------
__global__ __launch_bounds__(256) void softmax_rows(
    float* __restrict__ X, int nrows, int Lrow, float scale, int diagMod, float diagAdd)
{
  const int row = blockIdx.x * (blockDim.x >> 5) + (threadIdx.x >> 5);
  if (row >= nrows) return;
  const int lane = threadIdx.x & 31;
  float* x = X + (long)row * Lrow;
  const int dpos = (diagMod > 0) ? (row % diagMod) : -1;

  float vals[17];
  float vmax = -3.4e38f;
  int cnt = 0;
  for (int i = lane; i < Lrow; i += 32) {
    float v = x[i] * scale;
    if (i == dpos) v += diagAdd;
    vals[cnt++] = v;
    vmax = fmaxf(vmax, v);
  }
  for (int o = 16; o > 0; o >>= 1) vmax = fmaxf(vmax, __shfl_xor(vmax, o, 32));
  float s = 0.f;
  for (int j = 0; j < cnt; ++j) { vals[j] = __expf(vals[j] - vmax); s += vals[j]; }
  for (int o = 16; o > 0; o >>= 1) s += __shfl_xor(s, o, 32);
  const float inv = 1.0f / s;
  cnt = 0;
  for (int i = lane; i < Lrow; i += 32) x[i] = vals[cnt++] * inv;
}

// ---------------------------------------------------------------------------
// Elementwise / small kernels
// ---------------------------------------------------------------------------
__global__ void concat_sp_kernel(const float* __restrict__ samples,
                                 const float* __restrict__ relation,
                                 float* __restrict__ sp)
{
  using namespace cfg;
  long idx = (long)blockIdx.x * 256 + threadIdx.x;
  const long total = (long)B * LP * H2;
  if (idx >= total) return;
  const int c = (int)(idx % H2);
  const long t = idx / H2;
  const int r = (int)(t % LP);
  const int b = (int)(t / LP);
  float v = (r < L) ? samples[((long)b * L + r) * H2 + c]
                    : relation[((long)b * N + (r - L)) * H2 + c];
  sp[idx] = v;
}

__global__ void concat_out_kernel(const float* __restrict__ gat,
                                  const float* __restrict__ sp,
                                  float* __restrict__ sd, float* __restrict__ rd)
{
  using namespace cfg;
  long idx = (long)blockIdx.x * 256 + threadIdx.x;
  const long total = (long)B * LP * H3;
  if (idx >= total) return;
  const int c = (int)(idx % H3);
  const long t = idx / H3;
  const int r = (int)(t % LP);
  const int b = (int)(t / LP);
  float v = (c < H) ? gat[((long)b * LP + r) * H + c]
                    : sp[((long)b * LP + r) * H2 + (c - H)];
  if (r < L) sd[((long)b * L + r) * H3 + c] = v;
  else       rd[((long)b * N + (r - L)) * H3 + c] = v;
}

__global__ void diag_extract_kernel(const float* __restrict__ gram, float* __restrict__ x2)
{
  using namespace cfg;
  int idx = blockIdx.x * 256 + threadIdx.x;
  if (idx >= B * L) return;
  const int b = idx >> 9, l = idx & (L - 1);
  x2[idx] = gram[((long)b * L + l) * L + l];
}

__global__ void graph_finalize_kernel(float* __restrict__ gram, const float* __restrict__ x2)
{
  using namespace cfg;
  long idx = (long)blockIdx.x * 256 + threadIdx.x;
  const long total = (long)B * L * L;
  if (idx >= total) return;
  const int m = (int)(idx & (L - 1));
  const long t = idx >> 9;
  const int l = (int)(t & (L - 1));
  const int b = (int)(t >> 9);
  const float g = gram[idx];
  float d = fmaxf(x2[b * L + l] + x2[b * L + m] - 2.0f * g, 0.0f);
  float out = -sqrtf(d + 1e-6f);
  if (l == m) out += -1000000.0f;
  gram[idx] = out;
}

__global__ void copy_kernel(const float* __restrict__ src, float* __restrict__ dst, int n)
{
  int i = blockIdx.x * 256 + threadIdx.x;
  if (i < n) dst[i] = src[i];
}

__global__ void zero_kernel(float* __restrict__ dst, int n)
{
  int i = blockIdx.x * 256 + threadIdx.x;
  if (i < n) dst[i] = 0.0f;
}

// sample_w[b][l] = max_m ( graph*Mask + (1-Mask)*-1e6 ),  Mask = label.label^T
__global__ void maskmax_kernel(const float* __restrict__ graph,
                               const float* __restrict__ label,
                               float* __restrict__ samplew)
{
  using namespace cfg;
  int idx = blockIdx.x * 256 + threadIdx.x;
  if (idx >= B * L) return;
  const int b = idx >> 9, l = idx & (L - 1);
  float lab[N];
  const float* lb = label + ((long)b * L + l) * N;
#pragma unroll
  for (int n = 0; n < N; ++n) lab[n] = lb[n];
  const float* labB = label + (long)b * L * N;
  const float* gr = graph + ((long)b * L + l) * L;
  float w = -3.4e38f;
  for (int m = 0; m < L; ++m) {
    const float* lm = labB + (long)m * N;
    float mask = 0.f;
#pragma unroll
    for (int n = 0; n < N; ++n) mask += lab[n] * lm[n];
    float v = gr[m] * mask + (1.0f - mask) * (-1000000.0f);
    w = fmaxf(w, v);
  }
  samplew[idx] = w;
}

// lt1[b][n][l] = softmax_l( label[l][n]*w[l] + (1-label[l][n])*-1e6 )
__global__ __launch_bounds__(256) void lt1_kernel(const float* __restrict__ label,
                                                  const float* __restrict__ samplew,
                                                  float* __restrict__ lt1)
{
  using namespace cfg;
  const int row = blockIdx.x * (blockDim.x >> 5) + (threadIdx.x >> 5);
  if (row >= B * N) return;
  const int b = row >> 4, n = row & (N - 1);
  const int lane = threadIdx.x & 31;
  float vals[16];
  float vmax = -3.4e38f;
#pragma unroll
  for (int j = 0; j < 16; ++j) {
    const int l = lane + 32 * j;
    const float lb = label[((long)b * L + l) * N + n];
    const float v = lb * samplew[b * L + l] + (1.0f - lb) * (-1000000.0f);
    vals[j] = v;
    vmax = fmaxf(vmax, v);
  }
  for (int o = 16; o > 0; o >>= 1) vmax = fmaxf(vmax, __shfl_xor(vmax, o, 32));
  float s = 0.f;
#pragma unroll
  for (int j = 0; j < 16; ++j) { vals[j] = __expf(vals[j] - vmax); s += vals[j]; }
  for (int o = 16; o > 0; o >>= 1) s += __shfl_xor(s, o, 32);
  const float inv = 1.0f / s;
#pragma unroll
  for (int j = 0; j < 16; ++j) lt1[((long)b * N + n) * L + lane + 32 * j] = vals[j] * inv;
}

// Center[b][n][d] = 32/33 * sum_l lt1[n][l]*sd[l][d] + 1/33 * rd[n][d]
__global__ void center_kernel(const float* __restrict__ lt1,
                              const float* __restrict__ sd,
                              const float* __restrict__ rd,
                              float* __restrict__ center)
{
  using namespace cfg;
  long idx = (long)blockIdx.x * 256 + threadIdx.x;
  const long total = (long)B * N * H3;
  if (idx >= total) return;
  const int d = (int)(idx % H3);
  const long t = idx / H3;
  const int n = (int)(t & (N - 1));
  const int b = (int)(t >> 4);
  const float* w = lt1 + ((long)b * N + n) * L;
  const float* x = sd + (long)b * L * H3 + d;
  float acc = 0.f;
  for (int l = 0; l < L; ++l) acc += w[l] * x[(long)l * H3];
  center[idx] = (32.0f / 33.0f) * acc + (1.0f / 33.0f) * rd[((long)b * N + n) * H3 + d];
}

// Predict[b][l][n] = -||sd[l]-center[n]||^2 ; accumulate logits (mean over iters)
__global__ void predict_kernel(const float* __restrict__ sd,
                               const float* __restrict__ center,
                               float* __restrict__ pred,
                               float* __restrict__ logit, int iter)
{
  using namespace cfg;
  int idx = blockIdx.x * 256 + threadIdx.x;
  if (idx >= B * L * N) return;
  const int n = idx & (N - 1);
  const int t = idx >> 4;
  const int l = t & (L - 1);
  const int b = t >> 9;
  const float* x = sd + ((long)b * L + l) * H3;
  const float* c = center + ((long)b * N + n) * H3;
  float s = 0.f;
  for (int d = 0; d < H3; ++d) { const float df = x[d] - c[d]; s += df * df; }
  const float p = -s;
  pred[idx] = p;
  if (iter == 0) logit[idx] = p; else logit[idx] += p;
}

// _to_one_hot: one 512-thread block per batch; P (16x512) lives in LDS.
// kth-largest per row via monotone-uint bit binary search (tie-exact).
__global__ __launch_bounds__(512) void one_hot_kernel(const float* __restrict__ pred,
                                                      float* __restrict__ label, int Ntop)
{
  using namespace cfg;
  __shared__ float P[N][L + 16];
  __shared__ unsigned char M2[N][L + 16];
  __shared__ float thr[N];
  __shared__ int sAny;
  __shared__ int sDone;
  const int b = blockIdx.x;
  const int tid = threadIdx.x;           // column l
  const int wr = tid >> 5;               // wave = row n
  const int lane = tid & 31;
  const float* pb = pred + (long)b * L * N;
#pragma unroll
  for (int n = 0; n < N; ++n) P[n][tid] = pb[(long)tid * N + n];
  __syncthreads();

  for (int iter = 0;; ++iter) {
    // --- per-row kth-largest threshold (wave wr owns row wr) ---
    {
      unsigned int u[16];
#pragma unroll
      for (int j = 0; j < 16; ++j) {
        const unsigned int bits = __float_as_uint(P[wr][lane + 32 * j]);
        u[j] = (bits & 0x80000000u) ? ~bits : (bits | 0x80000000u);
      }
      unsigned int ans = 0u;
      for (int bit = 31; bit >= 0; --bit) {
        const unsigned int cand = ans | (1u << bit);
        int c = 0;
#pragma unroll
        for (int j = 0; j < 16; ++j) c += (u[j] >= cand) ? 1 : 0;
        for (int o = 16; o > 0; o >>= 1) c += __shfl_xor(c, o, 32);
        if (c >= Ntop) ans = cand;
      }
      if (lane == 0) {
        const unsigned int bits = (ans & 0x80000000u) ? (ans & 0x7FFFFFFFu) : ~ans;
        thr[wr] = __uint_as_float(bits);
      }
    }
    if (tid == 0) sAny = 0;
    __syncthreads();

    // --- per-column (sample) conflict resolution ---
    float p[N], m2[N], m4[N];
    int cs2 = 0;
#pragma unroll
    for (int n = 0; n < N; ++n) {
      p[n] = P[n][tid];
      m2[n] = (p[n] >= thr[n]) ? 1.f : 0.f;
      M2[n][tid] = (unsigned char)m2[n];
      cs2 += (int)m2[n];
    }
    if (cs2 >= 2) atomicOr(&sAny, 1);
    float vmax = -3.4e38f;
#pragma unroll
    for (int n = 0; n < N; ++n) vmax = fmaxf(vmax, p[n] - (1.f - m2[n]) * 1000000.0f);
    float col = 0.f;
#pragma unroll
    for (int n = 0; n < N; ++n) {
      const float p1 = p[n] - (1.f - m2[n]) * 1000000.0f;
      const float m3 = (p1 >= vmax) ? 1.f : 0.f;
      m4[n] = m3 * m2[n];
      col += m4[n];
    }
    const float m5 = (col > 0.5f) ? 1.f : 0.f;
    const float m6 = (col > 1.5f) ? 1.f : 0.f;
#pragma unroll
    for (int n = 0; n < N; ++n) {
      const float p2 = p[n] - m5 * 1000000.0f;
      const float mm = m4[n] * (1.f - m6);
      P[n][tid] = mm * p[n] + (1.f - mm) * p2;
    }
    __syncthreads();
    if (tid == 0) sDone = (sAny == 0) || (iter + 1 > 10);
    __syncthreads();
    if (sDone) break;
  }

  float* lb = label + (long)b * L * N;
#pragma unroll
  for (int n = 0; n < N; ++n) lb[(long)tid * N + n] = (float)M2[n][tid];
}

// Predict_in = softmax(pred/30) over the 16 classes
__global__ void predin_kernel(const float* __restrict__ pred, float* __restrict__ pi)
{
  using namespace cfg;
  int idx = blockIdx.x * 256 + threadIdx.x;
  if (idx >= B * L) return;
  const float* p = pred + (long)idx * N;
  float v[N];
  float mx = -3.4e38f;
#pragma unroll
  for (int n = 0; n < N; ++n) { v[n] = p[n] * (1.0f / 30.0f); mx = fmaxf(mx, v[n]); }
  float s = 0.f;
#pragma unroll
  for (int n = 0; n < N; ++n) { v[n] = __expf(v[n] - mx); s += v[n]; }
  const float inv = 1.0f / s;
#pragma unroll
  for (int n = 0; n < N; ++n) pi[(long)idx * N + n] = v[n] * inv;
}

__global__ void finalize_kernel(const float* __restrict__ pred,
                                float* __restrict__ outPred, float* __restrict__ outLogit)
{
  using namespace cfg;
  int i = blockIdx.x * 256 + threadIdx.x;
  if (i >= B * L * N) return;
  outPred[i] = pred[i];
  outLogit[i] *= 0.25f;
}

__global__ void dcalc_kernel(const float* __restrict__ edge, float* __restrict__ D)
{
  using namespace cfg;
  int idx = blockIdx.x * 256 + threadIdx.x;
  if (idx >= B * L) return;
  const float* e = edge + (long)idx * L;
  float s = 0.f;
  for (int m = 0; m < L; ++m) s += e[m];
  D[idx] = 1.0f / (s + 1e-10f);
}

// Pmat[b][l][m] = D[b][l] * edge[b][m][l]
__global__ void pmat_kernel(const float* __restrict__ edge, const float* __restrict__ D,
                            float* __restrict__ pmat)
{
  using namespace cfg;
  long idx = (long)blockIdx.x * 256 + threadIdx.x;
  const long total = (long)B * L * L;
  if (idx >= total) return;
  const int m = (int)(idx & (L - 1));
  const long t = idx >> 9;
  const int l = (int)(t & (L - 1));
  const int b = (int)(t >> 9);
  pmat[idx] = D[b * L + l] * edge[((long)b * L + m) * L + l];
}

// Db[b][l][n] = D[l] * (ps[l][n] - avg[n]) * has_lab[l]   (deterministic reductions)
__global__ __launch_bounds__(512) void bl_kernel(const float* __restrict__ label,
                                                 const float* __restrict__ D,
                                                 float* __restrict__ Db)
{
  using namespace cfg;
  __shared__ float red[512];
  const int b = blockIdx.x;
  const int tid = threadIdx.x;    // sample l
  const float* lb = label + ((long)b * L + tid) * N;
  float ps[N];
  float rs = 0.f;
#pragma unroll
  for (int n = 0; n < N; ++n) { ps[n] = lb[n]; rs += ps[n]; }
  const float has = (rs > 0.5f) ? 1.f : 0.f;

  red[tid] = has;
  __syncthreads();
  for (int s = 256; s > 0; s >>= 1) { if (tid < s) red[tid] += red[tid + s]; __syncthreads(); }
  const float total = red[0];
  __syncthreads();

  float cs[N];
  for (int n = 0; n < N; ++n) {
    red[tid] = ps[n];
    __syncthreads();
    for (int s = 256; s > 0; s >>= 1) { if (tid < s) red[tid] += red[tid + s]; __syncthreads(); }
    cs[n] = red[0];
    __syncthreads();
  }
  const float invT = 1.0f / total;
  const float d = D[b * L + tid];
  float* db = Db + ((long)b * L + tid) * N;
#pragma unroll
  for (int n = 0; n < N; ++n) db[n] = d * (ps[n] - cs[n] * invT) * has;
}

// ut_out[b][l][n] = sum_m Pmat[l][m]*ut_in[m][n] + Db[l][n] (+addv at t==3)
__global__ void ut_kernel(const float* __restrict__ pmat, const float* __restrict__ utin,
                          const float* __restrict__ Db, const float* __restrict__ addv,
                          float* __restrict__ utout)
{
  using namespace cfg;
  int idx = blockIdx.x * 256 + threadIdx.x;
  if (idx >= B * L * N) return;
  const int n = idx & (N - 1);
  const int t = idx >> 4;
  const int l = t & (L - 1);
  const int b = t >> 9;
  const float* pr = pmat + ((long)b * L + l) * L;
  const float* ui = utin + (long)b * L * N + n;
  float s = 0.f;
  for (int m = 0; m < L; ++m) s += pr[m] * ui[(long)m * N];
  s += Db[idx];
  if (addv) s += addv[idx];
  utout[idx] = s;
}

// ---------------------------------------------------------------------------
// Host-side helpers
// ---------------------------------------------------------------------------
static inline void launch_gemm(bool transb, const float* A, const float* Bm, const float* bias,
                               float* C, int M, int N, int K, int lda, int ldb, int ldc,
                               long sA, long hA, long sB, long hB, long sC, long hC,
                               int nh, int batches, hipStream_t st)
{
  dim3 grid((N + 63) / 64, (M + 127) / 128, batches);
  if (transb) {
    if (bias) wmma_gemm<true, true><<<grid, 256, 0, st>>>(A, Bm, bias, C, M, N, K, lda, ldb, ldc, sA, hA, sB, hB, sC, hC, nh);
    else      wmma_gemm<true, false><<<grid, 256, 0, st>>>(A, Bm, bias, C, M, N, K, lda, ldb, ldc, sA, hA, sB, hB, sC, hC, nh);
  } else {
    if (bias) wmma_gemm<false, true><<<grid, 256, 0, st>>>(A, Bm, bias, C, M, N, K, lda, ldb, ldc, sA, hA, sB, hB, sC, hC, nh);
    else      wmma_gemm<false, false><<<grid, 256, 0, st>>>(A, Bm, bias, C, M, N, K, lda, ldb, ldc, sA, hA, sB, hB, sC, hC, nh);
  }
}

static inline dim3 egrid(long n) { return dim3((unsigned)((n + 255) / 256)); }

extern "C" void kernel_launch(void* const* d_in, const int* in_sizes, int n_in,
                              void* d_out, int out_size, void* d_ws, size_t ws_size,
                              hipStream_t stream)
{
  using namespace cfg;
  (void)in_sizes; (void)n_in; (void)out_size; (void)ws_size;

  const float* samples  = (const float*)d_in[0];
  const float* relation = (const float*)d_in[1];
  const float* label_in = (const float*)d_in[2];
  const float* Wq = (const float*)d_in[3];  const float* bq = (const float*)d_in[4];
  const float* Wk = (const float*)d_in[5];  const float* bk = (const float*)d_in[6];
  const float* Wv = (const float*)d_in[7];  const float* bv = (const float*)d_in[8];
  const float* Wo = (const float*)d_in[9];  const float* bo = (const float*)d_in[10];
  const float* Wp = (const float*)d_in[11]; const float* bp = (const float*)d_in[12];

  float* out = (float*)d_out;
  float* outPred  = out;                                   // (B,L,N)
  float* outLogit = outPred + (long)B * L * N;             // (B,L,N)
  float* outSD    = outLogit + (long)B * L * N;            // (B,L,H3)
  float* outRD    = outSD + (long)B * L * H3;              // (B,N,H3)
  float* outLbl   = outRD + (long)B * N * H3;              // (B,L,N)

  // ---- workspace layout (fp32), with lifetime-based reuse ----
  float* ws   = (float*)d_ws;
  float* sp   = ws;                                  // B*LP*H2 = 25,952,256
  float* qb   = sp + (long)B * LP * H2;              // 12,976,128
  float* kb   = qb + (long)B * LP * H;               // 12,976,128
  float* vb   = kb + (long)B * LP * H;               // 12,976,128
  float* attn = vb + (long)B * LP * H;               // B*NH*LP*LP = 35,684,352
  float* ctx  = attn + (long)B * NH * LP * LP;       // 12,976,128
  float* gram = ctx + (long)B * LP * H;              // B*L*L = 8,388,608
  float* x2      = gram + (long)B * L * L;           // 16,384
  float* samplew = x2 + B * L;                       // 16,384
  float* labelb  = samplew + B * L;                  // 262,144
  float* lt1     = labelb + (long)B * L * N;         // 262,144
  float* center  = lt1 + (long)B * N * L;            // 1,179,648
  float* pred    = center + (long)B * N * H3;        // 262,144
  float* predin  = pred + (long)B * L * N;           // 262,144
  float* Dbuf    = predin + (long)B * L * N;         // 16,384
  float* Db      = Dbuf + B * L;                     // 262,144
  float* ut0     = Db + (long)B * L * N;             // 262,144
  float* ut1     = ut0 + (long)B * L * N;            // 262,144
  // reuse (dead regions):
  float* gat  = qb;                       // q dead after attention scores
  float* sbuf = kb;                       // k dead after attention scores
  float* edge = attn;                     // attn dead after ctx
  float* pmat = attn + (long)B * L * L;   // still inside attn region

  const long NROWS = (long)B * LP;        // 16896

  // ===== 1. sp = concat(samples, relation) =====
  concat_sp_kernel<<<egrid((long)B * LP * H2), 256, 0, stream>>>(samples, relation, sp);

  // ===== 2. q/k/v = sp @ W + b  (M=16896, K=1536, N=768) =====
  launch_gemm(false, sp, Wq, bq, qb, (int)NROWS, H, H2, H2, H, H, 0, 0, 0, 0, 0, 0, 1, 1, stream);
  launch_gemm(false, sp, Wk, bk, kb, (int)NROWS, H, H2, H2, H, H, 0, 0, 0, 0, 0, 0, 1, 1, stream);
  launch_gemm(false, sp, Wv, bv, vb, (int)NROWS, H, H2, H2, H, H, 0, 0, 0, 0, 0, 0, 1, 1, stream);

  // ===== 3. attention scores = q . k^T per (b,h)  (528x528, K=192) =====
  launch_gemm(true, qb, kb, nullptr, attn, LP, LP, DH, H, H, LP,
              (long)LP * H, (long)DH, (long)LP * H, (long)DH,
              (long)NH * LP * LP, (long)LP * LP, NH, B * NH, stream);

  // ===== 4. softmax over k =====
  softmax_rows<<<egrid((long)B * NH * LP * 32), 256, 0, stream>>>(attn, B * NH * LP, LP, 1.0f, 0, 0.0f);

  // ===== 5. ctx = attn @ v  (528x192, K=528) =====
  launch_gemm(false, attn, vb, nullptr, ctx, LP, DH, LP, LP, H, H,
              (long)NH * LP * LP, (long)LP * LP, (long)LP * H, (long)DH,
              (long)LP * H, (long)DH, NH, B * NH, stream);

  // ===== 6. gat = ctx @ Wo + bo =====
  launch_gemm(false, ctx, Wo, bo, gat, (int)NROWS, H, H, H, H, H, 0, 0, 0, 0, 0, 0, 1, 1, stream);

  // ===== 7. sd_full -> samples_d / relation_d outputs =====
  concat_out_kernel<<<egrid((long)B * LP * H3), 256, 0, stream>>>(gat, sp, outSD, outRD);

  // ===== 8-10. graph = -sqrt(sqdist + 1e-6) + eye*-1e6 =====
  launch_gemm(true, outSD, outSD, nullptr, gram, L, L, H3, H3, H3, L,
              (long)L * H3, 0, (long)L * H3, 0, (long)L * L, 0, 1, B, stream);
  diag_extract_kernel<<<egrid(B * L), 256, 0, stream>>>(gram, x2);
  graph_finalize_kernel<<<egrid((long)B * L * L), 256, 0, stream>>>(gram, x2);

  // ===== 11. init label =====
  copy_kernel<<<egrid(B * L * N), 256, 0, stream>>>(label_in, labelb, B * L * N);

  // ===== 12. k-means iterations =====
  for (int iter = 0; iter < 4; ++iter) {
    maskmax_kernel<<<egrid(B * L), 256, 0, stream>>>(gram, labelb, samplew);
    lt1_kernel<<<egrid((long)B * N * 32), 256, 0, stream>>>(labelb, samplew, lt1);
    center_kernel<<<egrid((long)B * N * H3), 256, 0, stream>>>(lt1, outSD, outRD, center);
    predict_kernel<<<egrid(B * L * N), 256, 0, stream>>>(outSD, center, pred, outLogit, iter);
    one_hot_kernel<<<dim3(B), 512, 0, stream>>>(pred, labelb, NTOP);
  }

  // ===== 13. final Predict / logits / Predict_in =====
  finalize_kernel<<<egrid(B * L * N), 256, 0, stream>>>(pred, outPred, outLogit);
  predin_kernel<<<egrid(B * L), 256, 0, stream>>>(pred, predin);

  // ===== 14. s = samples_d @ Wp + bp  (M=16384, K=2304, N=768) =====
  launch_gemm(false, outSD, Wp, bp, sbuf, B * L, H, H3, H3, H, H, 0, 0, 0, 0, 0, 0, 1, 1, stream);

  // ===== 15-16. edge = softmax(s.s^T + eye*-1e5) =====
  launch_gemm(true, sbuf, sbuf, nullptr, edge, L, L, H, H, H, L,
              (long)L * H, 0, (long)L * H, 0, (long)L * L, 0, 1, B, stream);
  softmax_rows<<<egrid((long)B * L * 32), 256, 0, stream>>>(edge, B * L, L, 1.0f, L, -100000.0f);

  // ===== 17-19. D, Pmat, Db =====
  dcalc_kernel<<<egrid(B * L), 256, 0, stream>>>(edge, Dbuf);
  pmat_kernel<<<egrid((long)B * L * L), 256, 0, stream>>>(edge, Dbuf, pmat);
  bl_kernel<<<dim3(B), 512, 0, stream>>>(labelb, Dbuf, Db);

  // ===== 20-21. diffusion: ut = P@ut + Db  (t=1..6, +Predict_in at t=3) =====
  zero_kernel<<<egrid(B * L * N), 256, 0, stream>>>(ut0, B * L * N);
  ut_kernel<<<egrid(B * L * N), 256, 0, stream>>>(pmat, ut0, Db, nullptr, ut1);   // t=1
  ut_kernel<<<egrid(B * L * N), 256, 0, stream>>>(pmat, ut1, Db, nullptr, ut0);   // t=2
  ut_kernel<<<egrid(B * L * N), 256, 0, stream>>>(pmat, ut0, Db, predin,  ut1);   // t=3
  ut_kernel<<<egrid(B * L * N), 256, 0, stream>>>(pmat, ut1, Db, nullptr, ut0);   // t=4
  ut_kernel<<<egrid(B * L * N), 256, 0, stream>>>(pmat, ut0, Db, nullptr, ut1);   // t=5
  ut_kernel<<<egrid(B * L * N), 256, 0, stream>>>(pmat, ut1, Db, nullptr, outLbl);// t=6
}
